// SelfAttention_16965120819886
// MI455X (gfx1250) — compile-verified, matrix operand order checked by reading, not run on hardware
//
#include <hip/hip_runtime.h>
#include <stdint.h>

typedef __attribute__((ext_vector_type(16))) __bf16 v16bf;
typedef __attribute__((ext_vector_type(8)))  float  v8f;
typedef __attribute__((ext_vector_type(8)))  unsigned short v8u16;

constexpr int Bsz = 4, Seq = 2048, Dim = 1024, Heads = 16, HDim = 64;
constexpr int Mrows = Bsz * Seq;  // 8192

__device__ __forceinline__ uint16_t f32_to_bf16(float f) {
  uint32_t u = __float_as_uint(f);
  u += 0x7FFFu + ((u >> 16) & 1u);   // round-to-nearest-even
  return (uint16_t)(u >> 16);
}

__device__ __forceinline__ v8f wmma_bf16(v16bf a, v16bf b, v8f c) {
  return __builtin_amdgcn_wmma_f32_16x16x32_bf16(false, a, false, b, (short)0, c,
                                                 false, false);
}

// Async global->LDS copy of 16 bytes per lane (CDNA5 ASYNCcnt path).
__device__ __forceinline__ void async_copy_b128(uint32_t lds_byte_off,
                                                const void* gptr) {
  asm volatile("global_load_async_to_lds_b128 %0, %1, off"
               :: "v"(lds_byte_off), "v"((uint64_t)(uintptr_t)gptr)
               : "memory");
}
__device__ __forceinline__ void wait_async0() {
  asm volatile("s_wait_asynccnt 0" ::: "memory");
}

// ---------------------------------------------------------------------------
// One-shot fp32 -> bf16 conversion (vectorized, memory-bound).
// ---------------------------------------------------------------------------
__global__ __launch_bounds__(256) void to_bf16(const float* __restrict__ in,
                                               uint16_t* __restrict__ out, int n) {
  const int i = (blockIdx.x * 256 + threadIdx.x) * 8;
  if (i >= n) return;            // n is always a multiple of 8
  const float4 f0 = *(const float4*)&in[i];
  const float4 f1 = *(const float4*)&in[i + 4];
  v8u16 o;
  o[0] = f32_to_bf16(f0.x); o[1] = f32_to_bf16(f0.y);
  o[2] = f32_to_bf16(f0.z); o[3] = f32_to_bf16(f0.w);
  o[4] = f32_to_bf16(f1.x); o[5] = f32_to_bf16(f1.y);
  o[6] = f32_to_bf16(f1.z); o[7] = f32_to_bf16(f1.w);
  *(v8u16*)&out[i] = o;
}

// ---------------------------------------------------------------------------
// GEMM: out[m,n] = sum_k A[m,k] * W[n,k] + bias[n]   (A @ W^T + b), bf16 in.
// Block tile 128x128, BK=64, 8 waves (2x4), each wave 64x32 (4x2 WMMA tiles).
// Staging uses async global->LDS b128 copies (no VGPR round trip).
// OUT_MODE 0: bf16 [Mrows,Dim] | 1: bf16 [B,H,HD,S] (V^T) | 2: fp32 [Mrows,Dim]
// ---------------------------------------------------------------------------
template <int OUT_MODE>
__global__ __launch_bounds__(256) void gemm_proj(const uint16_t* __restrict__ A,
                                                 const uint16_t* __restrict__ W,
                                                 const float* __restrict__ bias,
                                                 void* __restrict__ outp) {
  __shared__ __align__(32) uint16_t sA[128 * 64];   // 16 KB
  __shared__ __align__(32) uint16_t sB[128 * 64];   // 16 KB

  const int t    = threadIdx.x;
  const int wid  = t >> 5;
  const int lane = t & 31;
  const int r16  = lane & 15;
  const int hi   = lane >> 4;
  const int m_off = (wid & 1) * 64;
  const int n_off = (wid >> 1) * 32;
  const int mBase = blockIdx.x * 128;
  const int nBase = blockIdx.y * 128;

  const int ldRow = t >> 1;           // 0..127
  const int ldSeg = (t & 1) * 32;     // element offset 0 or 32 within 64-wide row

  v8f acc[4][2] = {};

  for (int k0 = 0; k0 < Dim; k0 += 64) {
    const uint16_t* aRow = &A[(size_t)(mBase + ldRow) * Dim + k0 + ldSeg];
    const uint16_t* wRow = &W[(size_t)(nBase + ldRow) * Dim + k0 + ldSeg];
    const uint32_t aOff = (uint32_t)(uintptr_t)&sA[ldRow * 64 + ldSeg];
    const uint32_t wOff = (uint32_t)(uintptr_t)&sB[ldRow * 64 + ldSeg];
#pragma unroll
    for (int c = 0; c < 4; ++c) {      // 4 x 16B = 64B per thread per tile
      async_copy_b128(aOff + c * 16, aRow + c * 8);
      async_copy_b128(wOff + c * 16, wRow + c * 8);
    }
    wait_async0();
    __syncthreads();

#pragma unroll
    for (int ks = 0; ks < 2; ++ks) {   // two K-slices of 32
      v16bf af[4], bf2[2];
#pragma unroll
      for (int mi = 0; mi < 4; ++mi)
        af[mi] = *(const v16bf*)&sA[(m_off + mi * 16 + r16) * 64 + ks * 32 + hi * 16];
#pragma unroll
      for (int ni = 0; ni < 2; ++ni)
        bf2[ni] = *(const v16bf*)&sB[(n_off + ni * 16 + r16) * 64 + ks * 32 + hi * 16];
#pragma unroll
      for (int mi = 0; mi < 4; ++mi)
#pragma unroll
        for (int ni = 0; ni < 2; ++ni)
          acc[mi][ni] = wmma_bf16(af[mi], bf2[ni], acc[mi][ni]);
    }
    __syncthreads();
  }

  // epilogue: C layout = VGPR r -> row (r + hi*8), lane r16 -> col
#pragma unroll
  for (int mi = 0; mi < 4; ++mi) {
#pragma unroll
    for (int ni = 0; ni < 2; ++ni) {
      const int n = nBase + n_off + ni * 16 + r16;
      const float bval = bias[n];
      const int mrow0 = mBase + m_off + mi * 16 + hi * 8;
#pragma unroll
      for (int r = 0; r < 8; ++r) {
        const int m = mrow0 + r;
        const float val = acc[mi][ni][r] + bval;
        if constexpr (OUT_MODE == 0) {
          ((uint16_t*)outp)[(size_t)m * Dim + n] = f32_to_bf16(val);
        } else if constexpr (OUT_MODE == 1) {
          const int bb = m >> 11, s = m & (Seq - 1);   // m = bb*Seq + s
          const int h = n >> 6, hd = n & 63;           // n = h*64 + hd
          ((uint16_t*)outp)[(((size_t)(bb * Heads + h)) * HDim + hd) * Seq + s] =
              f32_to_bf16(val);
        } else {
          ((float*)outp)[(size_t)m * Dim + n] = val;
        }
      }
    }
  }
}

// ---------------------------------------------------------------------------
// Flash attention: one wave per 16-query tile of one (b,h), 64-key KV tiles.
// Q,K bf16 [B*S, D]; Vt bf16 [B,H,HD,S]; X bf16 [B*S, D].
// scale folded to log2 domain: e * (1/32) * log2(e), probs via exp2.
// ---------------------------------------------------------------------------
__global__ __launch_bounds__(256) void flash_attn(const uint16_t* __restrict__ Q,
                                                  const uint16_t* __restrict__ K,
                                                  const uint16_t* __restrict__ Vt,
                                                  uint16_t* __restrict__ X) {
  __shared__ __align__(32) uint16_t sP[8][16 * 64];   // 2 KB per wave

  const int wid  = threadIdx.x >> 5;
  const int lane = threadIdx.x & 31;
  const int r16  = lane & 15;
  const int hi   = lane >> 4;
  const int gw = blockIdx.x * 8 + wid;
  const int mq = gw & 127;            // Seq/16 = 128 query tiles
  const int h  = (gw >> 7) & 15;
  const int b  = gw >> 11;

  const int qrow = b * Seq + mq * 16;
  const float scl2e = (1.0f / 32.0f) * 1.44269504f;   // 1/sqrt(D) * log2(e)

  const v16bf qf0 = *(const v16bf*)&Q[(size_t)(qrow + r16) * Dim + h * HDim + hi * 16];
  const v16bf qf1 =
      *(const v16bf*)&Q[(size_t)(qrow + r16) * Dim + h * HDim + 32 + hi * 16];

  v8f O[4] = {};
  float m_i[8], l_i[8];
#pragma unroll
  for (int r = 0; r < 8; ++r) { m_i[r] = -1e30f; l_i[r] = 0.0f; }

  const size_t kbase = (size_t)b * Seq * Dim + (size_t)h * HDim;
  const size_t vbase = ((size_t)(b * Heads + h)) * HDim * Seq;

  for (int j = 0; j < Seq / 64; ++j) {
    const int key0 = j * 64;
    v8f e[4] = {};
#pragma unroll
    for (int g = 0; g < 4; ++g) {      // 4 groups of 16 keys
      const size_t krow = kbase + (size_t)(key0 + g * 16 + r16) * Dim;
      v16bf kfa = *(const v16bf*)&K[krow + hi * 16];
      v16bf kfb = *(const v16bf*)&K[krow + 32 + hi * 16];
      e[g] = wmma_bf16(qf0, kfa, e[g]);
      e[g] = wmma_bf16(qf1, kfb, e[g]);
    }
    // online softmax (log2 domain); rows in VGPR index, keys across lanes
#pragma unroll
    for (int r = 0; r < 8; ++r) {
      float a0 = e[0][r] * scl2e, a1 = e[1][r] * scl2e;
      float a2 = e[2][r] * scl2e, a3 = e[3][r] * scl2e;
      float mx = fmaxf(fmaxf(a0, a1), fmaxf(a2, a3));
#pragma unroll
      for (int d = 1; d < 16; d <<= 1) mx = fmaxf(mx, __shfl_xor(mx, d, 16));
      const float mnew = fmaxf(m_i[r], mx);
      const float corr = exp2f(m_i[r] - mnew);
      const float p0 = exp2f(a0 - mnew), p1 = exp2f(a1 - mnew);
      const float p2 = exp2f(a2 - mnew), p3 = exp2f(a3 - mnew);
      float srow = (p0 + p1) + (p2 + p3);
#pragma unroll
      for (int d = 1; d < 16; d <<= 1) srow += __shfl_xor(srow, d, 16);
      l_i[r] = l_i[r] * corr + srow;
      m_i[r] = mnew;
#pragma unroll
      for (int tt = 0; tt < 4; ++tt) O[tt][r] *= corr;
      const int prow = (r + hi * 8) * 64;
      sP[wid][prow + r16]      = f32_to_bf16(p0);
      sP[wid][prow + r16 + 16] = f32_to_bf16(p1);
      sP[wid][prow + r16 + 32] = f32_to_bf16(p2);
      sP[wid][prow + r16 + 48] = f32_to_bf16(p3);
    }
    asm volatile("s_wait_dscnt 0" ::: "memory");      // LDS store->load turnaround
    const v16bf pf0 = *(const v16bf*)&sP[wid][r16 * 64 + hi * 16];
    const v16bf pf1 = *(const v16bf*)&sP[wid][r16 * 64 + 32 + hi * 16];
#pragma unroll
    for (int ni = 0; ni < 4; ++ni) {
      const size_t vrow = vbase + (size_t)(ni * 16 + r16) * Seq + key0;
      v16bf vfa = *(const v16bf*)&Vt[vrow + hi * 16];
      v16bf vfb = *(const v16bf*)&Vt[vrow + 32 + hi * 16];
      O[ni] = wmma_bf16(pf0, vfa, O[ni]);
      O[ni] = wmma_bf16(pf1, vfb, O[ni]);
    }
  }

  // normalize + store X in [B,S,D] (heads re-interleaved for final projection)
#pragma unroll
  for (int r = 0; r < 8; ++r) {
    const float inv = 1.0f / l_i[r];
    const int m = qrow + hi * 8 + r;
#pragma unroll
    for (int ni = 0; ni < 4; ++ni) {
      const int col = h * HDim + ni * 16 + r16;
      X[(size_t)m * Dim + col] = f32_to_bf16(O[ni][r] * inv);
    }
  }
}

// ---------------------------------------------------------------------------
extern "C" void kernel_launch(void* const* d_in, const int* in_sizes, int n_in,
                              void* d_out, int out_size, void* d_ws, size_t ws_size,
                              hipStream_t stream) {
  (void)in_sizes; (void)n_in; (void)out_size; (void)ws_size;
  const float* query = (const float*)d_in[0];
  const float* key   = (const float*)d_in[1];
  const float* value = (const float*)d_in[2];
  const float* Wq = (const float*)d_in[3];
  const float* bq = (const float*)d_in[4];
  const float* Wk = (const float*)d_in[5];
  const float* bk = (const float*)d_in[6];
  const float* Wv = (const float*)d_in[7];
  const float* bv = (const float*)d_in[8];
  const float* Wo = (const float*)d_in[9];
  const float* bo = (const float*)d_in[10];

  const size_t eA = (size_t)Mrows * Dim;   // 8.39M elems (16 MB bf16)
  const size_t eW = (size_t)Dim * Dim;     // 1.05M elems (2 MB bf16)
  uint16_t* aQ  = (uint16_t*)d_ws;         // bf16 activations
  uint16_t* aK  = aQ + eA;
  uint16_t* aV  = aK + eA;
  uint16_t* wQb = aV + eA;                 // bf16 weights
  uint16_t* wKb = wQb + eW;
  uint16_t* wVb = wKb + eW;
  uint16_t* wOb = wVb + eW;
  uint16_t* Qb  = wOb + eW;                // projected Q/K/V^T
  uint16_t* Kb  = Qb + eA;
  uint16_t* Vt  = Kb + eA;
  uint16_t* Xb  = aQ;                      // reuse: aQ dead after Q projection

  // 1) one-shot bf16 conversion (memory-bound pass)
  const int cvA = (int)(eA / 8 / 256), cvW = (int)(eW / 8 / 256);
  to_bf16<<<cvA, 256, 0, stream>>>(query, aQ, (int)eA);
  to_bf16<<<cvA, 256, 0, stream>>>(key,   aK, (int)eA);
  to_bf16<<<cvA, 256, 0, stream>>>(value, aV, (int)eA);
  to_bf16<<<cvW, 256, 0, stream>>>(Wq, wQb, (int)eW);
  to_bf16<<<cvW, 256, 0, stream>>>(Wk, wKb, (int)eW);
  to_bf16<<<cvW, 256, 0, stream>>>(Wv, wVb, (int)eW);
  to_bf16<<<cvW, 256, 0, stream>>>(Wo, wOb, (int)eW);

  // 2) Q/K/V projections (V stored transposed per head)
  const dim3 grid(Mrows / 128, Dim / 128);      // 64 x 8
  gemm_proj<0><<<grid, 256, 0, stream>>>(aQ, wQb, bq, Qb);
  gemm_proj<0><<<grid, 256, 0, stream>>>(aK, wKb, bk, Kb);
  gemm_proj<1><<<grid, 256, 0, stream>>>(aV, wVb, bv, Vt);

  // 3) flash attention
  const int waves = Bsz * Heads * (Seq / 16);   // 8192 waves
  flash_attn<<<dim3(waves / 8), 256, 0, stream>>>(Qb, Kb, Vt, Xb);

  // 4) output projection (fp32 out)
  gemm_proj<2><<<grid, 256, 0, stream>>>(Xb, wOb, bo, d_out);
}
